// TinyOnnMoE_2379411882358
// MI455X (gfx1250) — compile-verified, matrix operand order checked by reading, not run on hardware
//
#include <hip/hip_runtime.h>
#include <hip/hip_bf16.h>
#include <math.h>

// ---------------- problem constants ----------------
#define N_TOK 4096      // B*T
#define CDIM  1024
#define IDIM  2048
#define EEXP  16

// d_out float offsets (return-order concat)
#define OUT_FINAL 0
#define OUT_FULL  (N_TOK*CDIM)                          // 4194304
#define OUT_PRE   (OUT_FULL + (size_t)N_TOK*EEXP*CDIM)  // 71303168
#define OUT_MASK  (OUT_PRE + N_TOK*EEXP)                // 71368704

typedef __attribute__((ext_vector_type(16))) __bf16 v16bf;
typedef __attribute__((ext_vector_type(8)))  float  v8f;
typedef __attribute__((ext_vector_type(4)))  int    v4i;

union FragU { uint4 u[2]; v16bf v; };

#if __has_builtin(__builtin_amdgcn_global_load_async_to_lds_b128)
#define HAVE_ASYNC_LDS 1
typedef __attribute__((address_space(1))) v4i* gv4i_p;   // global
typedef __attribute__((address_space(3))) v4i* lv4i_p;   // LDS
__device__ __forceinline__ void async_copy_b128(const void* g, void* l) {
  __builtin_amdgcn_global_load_async_to_lds_b128((gv4i_p)g, (lv4i_p)l, 0, 0);
}
#endif

__device__ __forceinline__ void wait_asynccnt0() {
#if __has_builtin(__builtin_amdgcn_s_wait_asynccnt)
  __builtin_amdgcn_s_wait_asynccnt(0);
#else
  asm volatile("s_wait_asynccnt 0" ::: "memory");
#endif
}

__device__ __forceinline__ unsigned short bf16bits(float f) {
  __bf16 b = (__bf16)f;
  return __builtin_bit_cast(unsigned short, b);
}
__device__ __forceinline__ unsigned pack2(float a, float b) {
  return (unsigned)bf16bits(a) | ((unsigned)bf16bits(b) << 16);
}

// Branch-free exact-gelu: erf via Abramowitz-Stegun 7.1.26 (|err| < 1.5e-7)
__device__ __forceinline__ float gelu_exact(float v) {
  float x  = v * 0.70710678118654752f;
  float ax = fabsf(x);
  float t  = __builtin_amdgcn_rcpf(1.0f + 0.3275911f * ax);
  float p  = t * (0.254829592f +
             t * (-0.284496736f +
             t * (1.421413741f +
             t * (-1.453152027f +
             t * 1.061405429f))));
  float e  = __expf(-x * x);
  float er = copysignf(1.0f - p * e, x);
  return 0.5f * v * (1.0f + er);
}

// ---------------- x f32 -> bf16 ----------------
__global__ __launch_bounds__(256) void cvt_x(const float* __restrict__ x,
                                             unsigned short* __restrict__ xb) {
  int i = (blockIdx.x * 256 + threadIdx.x) * 4;   // N_TOK*CDIM divisible by 1024
  float4 v = *(const float4*)(x + i);
  uint2 o;
  o.x = pack2(v.x, v.y);
  o.y = pack2(v.z, v.w);
  *(uint2*)(xb + i) = o;
}

// ---------------- transpose+convert: W f32 [K][Nd] -> Wt bf16 [Nd][K] ----------------
__global__ __launch_bounds__(256) void transpose_w(const float* __restrict__ W,
                                                   unsigned short* __restrict__ Wt,
                                                   int K, int Nd) {
  const int kBase = blockIdx.y * 64;
  const int nBase = blockIdx.x * 64;
  const int tid = threadIdx.x;
  __shared__ __align__(16) float Tf[64 * 68];   // stride 68 to dodge bank conflicts

#pragma unroll
  for (int it = 0; it < 4; ++it) {
    int flat = tid + it * 256;         // 1024 float4 chunks
    int kk = flat >> 4;
    int nn = (flat & 15) << 2;
    float4 v = *(const float4*)(W + (size_t)(kBase + kk) * Nd + nBase + nn);
    Tf[kk * 68 + nn + 0] = v.x;
    Tf[kk * 68 + nn + 1] = v.y;
    Tf[kk * 68 + nn + 2] = v.z;
    Tf[kk * 68 + nn + 3] = v.w;
  }
  __syncthreads();

#pragma unroll
  for (int it = 0; it < 2; ++it) {
    int flat = tid + it * 256;         // 512 chunks of 8 k-elements
    int n  = flat >> 3;
    int kc = (flat & 7) << 3;
    uint4 o;
    o.x = pack2(Tf[(kc + 0) * 68 + n], Tf[(kc + 1) * 68 + n]);
    o.y = pack2(Tf[(kc + 2) * 68 + n], Tf[(kc + 3) * 68 + n]);
    o.z = pack2(Tf[(kc + 4) * 68 + n], Tf[(kc + 5) * 68 + n]);
    o.w = pack2(Tf[(kc + 6) * 68 + n], Tf[(kc + 7) * 68 + n]);
    *(uint4*)(Wt + (size_t)(nBase + n) * K + kBase + kc) = o;
  }
}

// ---------------- gate: l2norm logits, threshold, softmax routing ----------------
__global__ __launch_bounds__(256) void gate_kernel(const float* __restrict__ x,
                                                   const float* __restrict__ sim,
                                                   const float* __restrict__ gates,
                                                   float* __restrict__ pre_out,
                                                   float* __restrict__ mask_out,
                                                   float* __restrict__ rw_out) {
  int t = blockIdx.x * 256 + threadIdx.x;  // [0, N_TOK*EEXP)
  int tok = t >> 4;
  int e   = t & 15;
  const float* xr = x + (size_t)tok * CDIM;
  float dot = 0.f, xs = 0.f, ss = 0.f;
  for (int c = 0; c < CDIM; ++c) {
    float xc = xr[c];
    float sc = sim[c * EEXP + e];
    dot += xc * sc;
    xs  += xc * xc;
    ss  += sc * sc;
  }
  const float eps = 1e-12f;
  float denom = fmaxf(sqrtf(xs), eps) * fmaxf(sqrtf(ss), eps);
  float logit = dot / denom;
  float th = 1.0f / (1.0f + expf(-gates[e]));
  float pre = logit - th;
  float gated = fmaxf(pre, 0.0f);
  float mk = (gated > 0.0f) ? 1.0f : 0.0f;
  float masked = (mk > 0.0f) ? gated : -3.402823466e+38f;
  float m = masked;
  for (int off = 8; off > 0; off >>= 1) m = fmaxf(m, __shfl_xor(m, off, 16));
  float ex = expf(masked - m);
  float s = ex;
  for (int off = 8; off > 0; off >>= 1) s += __shfl_xor(s, off, 16);
  pre_out[t]  = pre;
  mask_out[t] = mk;
  rw_out[t]   = ex / s;
}

// ---------------- tiled bf16 WMMA GEMM (both operands bf16, K-major) ----------------
// Macro tile 256(M) x 128(N) x 32(K); 8 waves, each owns 64x64 (4x4 WMMA tiles).
// MODE 0: Hout[r,c]   = bf16(gelu(acc))    (Nd = IDIM)
// MODE 1: Fout[r,e,c] = mask[r,e]*acc      (Nd = CDIM)
template <int MODE>
__global__ __launch_bounds__(256)
void gemm_wmma(const unsigned short* __restrict__ A,   // bf16 [4096 x K]
               const unsigned short* __restrict__ Bt,  // bf16 [Nd x K]
               int K, int Nd,
               unsigned short* __restrict__ Hout,
               float* __restrict__ Fout,
               const float* __restrict__ mask,
               int expert) {
  const int mBase = blockIdx.y * 256;
  const int nBase = blockIdx.x * 128;
  const int tid  = threadIdx.x;
  const int lane = tid & 31;
  const int wave = tid >> 5;
  const int wm = wave >> 1;   // 0..3 : rows [wm*64, +64)
  const int wn = wave & 1;    // 0..1 : cols [wn*64, +64)
  const int h  = lane >> 4;
  const int ml = lane & 15;

  __shared__ __align__(16) unsigned short As[256 * 40];  // [row][k], stride 40
  __shared__ __align__(16) unsigned short Bs[128 * 40];  // [col][k], stride 40

  v8f acc[4][4];
#pragma unroll
  for (int mi = 0; mi < 4; ++mi)
#pragma unroll
    for (int ni = 0; ni < 4; ++ni) acc[mi][ni] = (v8f)0.0f;

  for (int k0 = 0; k0 < K; k0 += 32) {
    // stage A: 256x32 bf16 = 1024 chunks of 8 ; stage Bt: 128x32 = 512 chunks
#ifdef HAVE_ASYNC_LDS
#pragma unroll
    for (int it = 0; it < 4; ++it) {
      int flat = tid + it * 256;
      int row = flat >> 2;
      int kk  = (flat & 3) << 3;
      async_copy_b128(A + (size_t)(mBase + row) * K + k0 + kk,
                      &As[row * 40 + kk]);
    }
#pragma unroll
    for (int it = 0; it < 2; ++it) {
      int flat = tid + it * 256;
      int row = flat >> 2;
      int kk  = (flat & 3) << 3;
      async_copy_b128(Bt + (size_t)(nBase + row) * K + k0 + kk,
                      &Bs[row * 40 + kk]);
    }
    wait_asynccnt0();
#else
#pragma unroll
    for (int it = 0; it < 4; ++it) {
      int flat = tid + it * 256;
      int row = flat >> 2;
      int kk  = (flat & 3) << 3;
      uint4 d = *(const uint4*)(A + (size_t)(mBase + row) * K + k0 + kk);
      *(uint4*)(&As[row * 40 + kk]) = d;
    }
#pragma unroll
    for (int it = 0; it < 2; ++it) {
      int flat = tid + it * 256;
      int row = flat >> 2;
      int kk  = (flat & 3) << 3;
      uint4 d = *(const uint4*)(Bt + (size_t)(nBase + row) * K + k0 + kk);
      *(uint4*)(&Bs[row * 40 + kk]) = d;
    }
#endif
    __syncthreads();

    FragU a[4], b[4];
#pragma unroll
    for (int mi = 0; mi < 4; ++mi) {
      int r = wm * 64 + mi * 16 + ml;
      a[mi].u[0] = *(const uint4*)(&As[r * 40 + h * 8]);       // K 8h..8h+7
      a[mi].u[1] = *(const uint4*)(&As[r * 40 + 16 + h * 8]);  // K 16+8h..+7
    }
#pragma unroll
    for (int ni = 0; ni < 4; ++ni) {
      int c = wn * 64 + ni * 16 + ml;
      b[ni].u[0] = *(const uint4*)(&Bs[c * 40 + h * 8]);
      b[ni].u[1] = *(const uint4*)(&Bs[c * 40 + 16 + h * 8]);
    }
#pragma unroll
    for (int mi = 0; mi < 4; ++mi)
#pragma unroll
      for (int ni = 0; ni < 4; ++ni)
        acc[mi][ni] = __builtin_amdgcn_wmma_f32_16x16x32_bf16(
            false, a[mi].v, false, b[ni].v, (short)0, acc[mi][ni], false, false);
    __syncthreads();
  }

  // epilogue: VGPR j -> M = j + 8*h ; N = lane%16
#pragma unroll
  for (int mi = 0; mi < 4; ++mi) {
#pragma unroll
    for (int j = 0; j < 8; ++j) {
      int r = mBase + wm * 64 + mi * 16 + j + 8 * h;
      float mk = (MODE == 1) ? mask[r * EEXP + expert] : 0.0f;
#pragma unroll
      for (int ni = 0; ni < 4; ++ni) {
        int c = nBase + wn * 64 + ni * 16 + ml;
        float v = acc[mi][ni][j];
        if (MODE == 0) {
          Hout[(size_t)r * Nd + c] = bf16bits(gelu_exact(v));
        } else {
          Fout[((size_t)r * EEXP + expert) * CDIM + c] = mk * v;
        }
      }
    }
  }
}

// ---------------- combine: final = sum_e rw * full ----------------
__global__ __launch_bounds__(256) void combine(const float* __restrict__ full,
                                               const float* __restrict__ rw,
                                               float* __restrict__ fin) {
  int t = blockIdx.x * 256 + threadIdx.x;
  int n = t >> 8;
  int c = (t & 255) << 2;
  float4 s = {0.f, 0.f, 0.f, 0.f};
#pragma unroll
  for (int e = 0; e < EEXP; ++e) {
    float w = rw[n * EEXP + e];
    float4 v = *(const float4*)(full + ((size_t)n * EEXP + e) * CDIM + c);
    s.x += w * v.x; s.y += w * v.y; s.z += w * v.z; s.w += w * v.w;
  }
  *(float4*)(fin + (size_t)n * CDIM + c) = s;
}

extern "C" void kernel_launch(void* const* d_in, const int* in_sizes, int n_in,
                              void* d_out, int out_size, void* d_ws, size_t ws_size,
                              hipStream_t stream) {
  const float* x     = (const float*)d_in[0];   // [N, C]
  const float* sim   = (const float*)d_in[1];   // [C, E]
  const float* gates = (const float*)d_in[2];   // [E]
  const float* w1    = (const float*)d_in[3];   // [E, C, I]
  const float* w2    = (const float*)d_in[4];   // [E, I, C]

  float* out  = (float*)d_out;
  float* fin  = out + OUT_FINAL;
  float* full = out + OUT_FULL;
  float* pre  = out + OUT_PRE;
  float* msk  = out + OUT_MASK;

  // ws layout: xb bf16 8MB | hb bf16 16MB | rw f32 .25MB | w1t bf16 4MB | w2t bf16 4MB
  unsigned short* xb  = (unsigned short*)d_ws;
  unsigned short* hb  = (unsigned short*)((char*)d_ws + (8u  << 20));
  float*          rw  = (float*)         ((char*)d_ws + (24u << 20));
  unsigned short* w1t = (unsigned short*)((char*)d_ws + (25u << 20));
  unsigned short* w2t = (unsigned short*)((char*)d_ws + (29u << 20));

  cvt_x<<<(N_TOK * CDIM) / 1024, 256, 0, stream>>>(x, xb);
  gate_kernel<<<(N_TOK * EEXP) / 256, 256, 0, stream>>>(x, sim, gates, pre, msk, rw);

  for (int e = 0; e < EEXP; ++e) {
    const float* w1e = w1 + (size_t)e * CDIM * IDIM;   // [C][I]
    const float* w2e = w2 + (size_t)e * IDIM * CDIM;   // [I][C]

    // w1t [I][C] bf16 ; H = gelu(X @ W1[e]) : M=4096, Nd=IDIM, K=CDIM
    transpose_w<<<dim3(IDIM / 64, CDIM / 64), 256, 0, stream>>>(w1e, w1t, CDIM, IDIM);
    gemm_wmma<0><<<dim3(IDIM / 128, N_TOK / 256), 256, 0, stream>>>(
        xb, w1t, CDIM, IDIM, hb, nullptr, nullptr, 0);

    // w2t [C][I] bf16 ; full[:,e,:] = mask * (H @ W2[e]) : M=4096, Nd=CDIM, K=IDIM
    transpose_w<<<dim3(CDIM / 64, IDIM / 64), 256, 0, stream>>>(w2e, w2t, IDIM, CDIM);
    gemm_wmma<1><<<dim3(CDIM / 128, N_TOK / 256), 256, 0, stream>>>(
        hb, w2t, IDIM, CDIM, nullptr, full, msk, e);
  }

  combine<<<N_TOK, 256, 0, stream>>>(full, rw, fin);
}
